// HabitatGraph_32847909880497
// MI455X (gfx1250) — matrix operand order repaired and checked by reading, not compiled, not run on hardware
//
#include <hip/hip_runtime.h>
#include <hip/hip_bf16.h>
#include <math.h>

// ---------------------------------------------------------------------------
// HabitatGraph edge-weight kernel for MI455X (gfx1250, wave32, WMMA)
//   B=256, H=256, FEAT=512
//   out[b,i,j] = m_und ? max(cos_sim(x_i,x_j),0) * exp(-w^2/(sigma^2+EPS)) : 0
// ---------------------------------------------------------------------------

typedef __attribute__((ext_vector_type(16))) _Float16 v16h;
typedef __attribute__((ext_vector_type(8)))  _Float16 v8h;
typedef __attribute__((ext_vector_type(4)))  _Float16 v4h;
typedef __attribute__((ext_vector_type(8)))  float    v8f;

#define B_   256
#define H_   256
#define F_   512
#define EPS_ 1e-6f

// ---- kernel 0: zero the global stats accumulators -------------------------
__global__ void hg_zero_stats(float* __restrict__ stats) {
  if (threadIdx.x < 4) stats[threadIdx.x] = 0.0f;
}

// ---- kernel 1: row-normalize x and convert f32 -> f16 ---------------------
// One wave per row of 512 floats; 8 rows per 256-thread block.
__global__ void hg_normalize(const float* __restrict__ x,
                             _Float16* __restrict__ xn) {
  const int wave = threadIdx.x >> 5;
  const int lane = threadIdx.x & 31;
  const int row  = blockIdx.x * 8 + wave;

  const float*  xr = x  + (size_t)row * F_;
  _Float16*     xo = xn + (size_t)row * F_;

  float4 v[4];
  float  sq = 0.0f;
#pragma unroll
  for (int c = 0; c < 4; ++c) {
    v[c] = ((const float4*)xr)[lane + 32 * c];           // coalesced 16B loads
    sq += v[c].x * v[c].x + v[c].y * v[c].y + v[c].z * v[c].z + v[c].w * v[c].w;
  }
#pragma unroll
  for (int off = 16; off > 0; off >>= 1)                 // wave32 reduction
    sq += __shfl_xor(sq, off, 32);

  const float s = rsqrtf(fmaxf(sq, 1e-24f));             // F.normalize eps
#pragma unroll
  for (int c = 0; c < 4; ++c) {
    v4h h;
    h[0] = (_Float16)(v[c].x * s);
    h[1] = (_Float16)(v[c].y * s);
    h[2] = (_Float16)(v[c].z * s);
    h[3] = (_Float16)(v[c].w * s);
    ((v4h*)xo)[lane + 32 * c] = h;                       // coalesced 8B stores
  }
}

// ---- kernel 2: global edge statistics (n, sum w, sum w^2) -----------------
// 16 elements per thread, wave-shuffle + LDS reduce, one atomicAdd per block.
__global__ void hg_edge_stats(const float* __restrict__ dist,
                              const int*   __restrict__ mask,
                              float* __restrict__ stats) {
  const int base = blockIdx.x * (256 * 16);
  float cnt = 0.0f, sw = 0.0f, sw2 = 0.0f;

#pragma unroll
  for (int k = 0; k < 16; ++k) {
    const int idx = base + k * 256 + threadIdx.x;        // coalesced
    const int b = idx >> 16;                             // H*H = 65536
    const int r = (idx >> 8) & 255;
    const int c = idx & 255;
    const float dij = dist[idx];
    const float dji = dist[(b << 16) | (c << 8) | r];
    const bool mi = mask[(b << 8) | r] != 0;
    const bool mj = mask[(b << 8) | c] != 0;
    const bool pe  = (r != c) && mi && mj;
    const bool eij = pe && (dij >= 0.0f);
    const bool eji = pe && (dji >= 0.0f);
    const bool mu  = eij || eji;
    const float w = (eij && eji) ? 0.5f * (dij + dji)
                                 : (eij ? dij : (eji ? dji : 0.0f));
    if (mu) { cnt += 1.0f; sw += w; sw2 += w * w; }
  }

#pragma unroll
  for (int off = 16; off > 0; off >>= 1) {
    cnt += __shfl_down(cnt, off, 32);
    sw  += __shfl_down(sw,  off, 32);
    sw2 += __shfl_down(sw2, off, 32);
  }

  __shared__ float sc[8], ss[8], s2[8];
  const int wave = threadIdx.x >> 5;
  const int lane = threadIdx.x & 31;
  if (lane == 0) { sc[wave] = cnt; ss[wave] = sw; s2[wave] = sw2; }
  __syncthreads();
  if (threadIdx.x == 0) {
    float C = 0.0f, S = 0.0f, S2 = 0.0f;
#pragma unroll
    for (int w = 0; w < 8; ++w) { C += sc[w]; S += ss[w]; S2 += s2[w]; }
    atomicAdd(&stats[0], C);
    atomicAdd(&stats[1], S);
    atomicAdd(&stats[2], S2);
  }
}

// ---- kernel 3: WMMA GEMM (sim = xn * xn^T per batch) + fused epilogue -----
// Grid: 8 blocks per batch * 256 batches. 256 threads = 8 waves per block.
// Each wave computes one 32x32 tile: 4 v_wmma_f32_16x16x32_f16 accumulators,
// K-loop over 512 in steps of 32.
__global__ void __launch_bounds__(256)
hg_gemm_epilogue(const _Float16* __restrict__ xn,
                 const float*    __restrict__ dist,
                 const int*      __restrict__ mask,
                 const float*    __restrict__ stats,
                 float* __restrict__ out) {
  const int wave  = threadIdx.x >> 5;
  const int lane  = threadIdx.x & 31;
  const int lr    = lane & 15;          // lane-in-group: M (A) / N (B)
  const int hh    = lane >> 4;          // lane half: selects K sub-range
  const int batch = blockIdx.x >> 3;
  const int m0    = (blockIdx.x & 7) * 32;   // output row tile
  const int n0    = wave * 32;               // output col tile

  const _Float16* Xb = xn + (size_t)batch * H_ * F_;
  const _Float16* Ar0 = Xb + (size_t)(m0 + lr)      * F_;
  const _Float16* Ar1 = Xb + (size_t)(m0 + 16 + lr) * F_;
  const _Float16* Br0 = Xb + (size_t)(n0 + lr)      * F_;
  const _Float16* Br1 = Xb + (size_t)(n0 + 16 + lr) * F_;

  v8f acc00 = {}, acc01 = {}, acc10 = {}, acc11 = {};

  union AFrag { v16h v; v8h h[2]; };

  for (int kk = 0; kk < F_; kk += 32) {
    // A fragment (16x32, row M=lr): two contiguous 8-half (16B) chunks.
    // elements 0..7 -> K = kk+8*hh .. +7 ; elements 8..15 -> K = kk+16+8*hh ..
    AFrag a0, a1;
    a0.h[0] = *(const v8h*)(Ar0 + kk + 8 * hh);
    a0.h[1] = *(const v8h*)(Ar0 + kk + 16 + 8 * hh);
    a1.h[0] = *(const v8h*)(Ar1 + kk + 8 * hh);
    a1.h[1] = *(const v8h*)(Ar1 + kk + 16 + 8 * hh);
    // B fragment (32x16, col N=lr): B = xn^T, same K order as A:
    // elements 0..7 -> K = kk+8*hh .. +7 ; elements 8..15 -> K = kk+16+8*hh ..
    AFrag bf0, bf1;
    bf0.h[0] = *(const v8h*)(Br0 + kk + 8 * hh);
    bf0.h[1] = *(const v8h*)(Br0 + kk + 16 + 8 * hh);
    bf1.h[0] = *(const v8h*)(Br1 + kk + 8 * hh);
    bf1.h[1] = *(const v8h*)(Br1 + kk + 16 + 8 * hh);
    const v16h b0 = bf0.v;
    const v16h b1 = bf1.v;

    acc00 = __builtin_amdgcn_wmma_f32_16x16x32_f16(false, a0.v, false, b0,
                                                   (short)0, acc00, false, false);
    acc01 = __builtin_amdgcn_wmma_f32_16x16x32_f16(false, a0.v, false, b1,
                                                   (short)0, acc01, false, false);
    acc10 = __builtin_amdgcn_wmma_f32_16x16x32_f16(false, a1.v, false, b0,
                                                   (short)0, acc10, false, false);
    acc11 = __builtin_amdgcn_wmma_f32_16x16x32_f16(false, a1.v, false, b1,
                                                   (short)0, acc11, false, false);
  }

  // sigma from global stats (torch unbiased std), then fused edge weighting.
  const float n    = stats[0];
  const float mean = stats[1] / fmaxf(n, 1.0f);
  const float var  = fmaxf(stats[2] - n * mean * mean, 0.0f) / fmaxf(n - 1.0f, 1.0f);
  const float sigma = fmaxf(sqrtf(var), EPS_);
  const float inv   = 1.0f / (sigma * sigma + EPS_);

  const float* Db = dist + (size_t)batch * H_ * H_;
  const int*   Mb = mask + batch * H_;
  float*       Ob = out  + (size_t)batch * H_ * H_;

  v8f accs[2][2] = { { acc00, acc01 }, { acc10, acc11 } };

#pragma unroll
  for (int sm = 0; sm < 2; ++sm) {
#pragma unroll
    for (int sn = 0; sn < 2; ++sn) {
      const v8f a = accs[sm][sn];
      const int col = n0 + sn * 16 + lr;
      const bool mj = Mb[col] != 0;
#pragma unroll
      for (int i = 0; i < 8; ++i) {
        // C/D layout: VGPR i, lanes 0-15 -> M = i, lanes 16-31 -> M = 8+i
        const int row = m0 + sm * 16 + hh * 8 + i;
        const float dij = Db[row * H_ + col];
        const float dji = Db[col * H_ + row];
        const bool mi  = Mb[row] != 0;
        const bool pe  = (row != col) && mi && mj;
        const bool eij = pe && (dij >= 0.0f);
        const bool eji = pe && (dji >= 0.0f);
        const bool mu  = eij || eji;
        const float w  = (eij && eji) ? 0.5f * (dij + dji)
                                      : (eij ? dij : (eji ? dji : 0.0f));
        const float ew = fmaxf(a[i], 0.0f) * __expf(-(w * w) * inv);
        Ob[row * H_ + col] = mu ? ew : 0.0f;
      }
    }
  }
}

// ---------------------------------------------------------------------------
extern "C" void kernel_launch(void* const* d_in, const int* in_sizes, int n_in,
                              void* d_out, int out_size, void* d_ws, size_t ws_size,
                              hipStream_t stream) {
  (void)in_sizes; (void)n_in; (void)out_size; (void)ws_size;

  const float* x    = (const float*)d_in[0];   // [B*H, FEAT] f32
  const float* dist = (const float*)d_in[1];   // [B, H, H]   f32
  const int*   mask = (const int*)d_in[2];     // [B, H]      bool->int
  float*       out  = (float*)d_out;           // [B, H, H]   f32

  float*    stats = (float*)d_ws;                        // 3 accumulators
  _Float16* xn    = (_Float16*)((char*)d_ws + 256);      // 64 MB f16 rows

  hg_zero_stats   <<<1,    32,  0, stream>>>(stats);
  hg_normalize    <<<8192, 256, 0, stream>>>(x, xn);     // 65536 rows / 8
  hg_edge_stats   <<<4096, 256, 0, stream>>>(dist, mask, stats);
  hg_gemm_epilogue<<<2048, 256, 0, stream>>>(xn, dist, mask, stats, out);
}
